// HGT_3298534884299
// MI455X (gfx1250) — matrix operand order repaired and checked by reading, not compiled, not run on hardware
//
#include <hip/hip_runtime.h>
#include <hip/hip_bf16.h>
#include <math.h>

// ---------------------------------------------------------------------------
// CDNA5 (gfx1250) HGT forward.  wave32, WMMA bf16 16x16x32 w/ f32 accumulate.
// ---------------------------------------------------------------------------

typedef __attribute__((ext_vector_type(16))) __bf16 v16bf;
typedef __attribute__((ext_vector_type(8)))  float  v8f;

#define N_NODES 50000
#define F_IN_   768
#define HID     128
#define N_EDGES 400000

// ---------------------------------------------------------------------------
// Weight packing: W[K x N] f32 (row-major)  ->  fragment-major bf16.
// Packed element index:
//   (( nTile16 * (K/32) + kChunk32 ) * 32 + lane ) * 16 + i
// holds W[ kChunk32*32 + i + 16*(lane>>4) ][ nTile16*16 + (lane&15) ].
// In the GEMM each lane then loads its whole v16bf B fragment as one
// contiguous 32-byte (2 x b128) load.
// ---------------------------------------------------------------------------
__global__ void pack_w_kernel(const float* __restrict__ W, __bf16* __restrict__ P,
                              int K, int N)
{
    int t = (int)(blockIdx.x * blockDim.x + threadIdx.x);
    if (t >= K * N) return;
    int i    = t & 15;
    int lane = (t >> 4) & 31;
    int ct   = t >> 9;
    int kch  = K >> 5;
    int c    = ct % kch;           // k chunk (32 rows)
    int nt   = ct / kch;           // n tile (16 cols)
    int grp  = lane >> 4;
    int kk   = c * 32 + i + (grp << 4);
    int col  = nt * 16 + (lane & 15);
    P[t] = (__bf16)W[(size_t)kk * N + col];
}

// ---------------------------------------------------------------------------
// WMMA GEMM:  C[M x N] = post( A[M x K] * Wpacked[K x N] + bias )
// One wave computes a 16x64 tile (4 WMMAs / K-step sharing one A fragment).
// Requires M%16==0, N%64==0, K == KCH*32 (compile-time trip count so the
// K-loop unrolls and loads pipeline across WMMAs).
//   POST: 0 = none, 1 = LeakyReLU(0.01), 2 = skip-mix:
//         beta = sigmoid(skip[0]); C = beta*val + (1-beta)*residual
// ---------------------------------------------------------------------------
template <int POST, int KCH>
__global__ __launch_bounds__(256) void wmma_gemm_kernel(
    const float* __restrict__ A, const __bf16* __restrict__ P,
    const float* __restrict__ bias, float* __restrict__ C,
    int M, int N,
    const float* __restrict__ residual, const float* __restrict__ skipPtr)
{
    const int K = KCH * 32;
    int wave = (int)((blockIdx.x * blockDim.x + threadIdx.x) >> 5);
    int lane = (int)(threadIdx.x & 31);
    int nt64   = N >> 6;
    int tiles  = (M >> 4) * nt64;
    if (wave >= tiles) return;              // wave-uniform: EXEC stays all-1s

    int m0   = (wave / nt64) << 4;
    int nt0  = (wave % nt64) << 2;          // first 16-col tile index (4 tiles)
    int row  = lane & 15;
    int grp  = lane >> 4;

    v8f acc0 = {}, acc1 = {}, acc2 = {}, acc3 = {};
    const float* arow = A + (size_t)(m0 + row) * (size_t)K;
    const __bf16* pb0 = P + ((((size_t)(nt0 + 0) * KCH) * 32 + lane) << 4);
    const __bf16* pb1 = P + ((((size_t)(nt0 + 1) * KCH) * 32 + lane) << 4);
    const __bf16* pb2 = P + ((((size_t)(nt0 + 2) * KCH) * 32 + lane) << 4);
    const __bf16* pb3 = P + ((((size_t)(nt0 + 3) * KCH) * 32 + lane) << 4);

#pragma unroll
    for (int c = 0; c < KCH; ++c) {
        int k0 = c << 5;
        // ---- A fragment: one contiguous 16-float run per lane -> 4 x b128
        const float4* apLo = (const float4*)(arow + k0 + (grp << 4));
        const float4* apHi = (const float4*)(arow + k0 + 8 + (grp << 4));
        float4 f0 = apLo[0], f1 = apLo[1];
        float4 f2 = apHi[0], f3 = apHi[1];
        v16bf a;
        a[0]  = (__bf16)f0.x; a[1]  = (__bf16)f0.y; a[2]  = (__bf16)f0.z; a[3]  = (__bf16)f0.w;
        a[4]  = (__bf16)f1.x; a[5]  = (__bf16)f1.y; a[6]  = (__bf16)f1.z; a[7]  = (__bf16)f1.w;
        a[8]  = (__bf16)f2.x; a[9]  = (__bf16)f2.y; a[10] = (__bf16)f2.z; a[11] = (__bf16)f2.w;
        a[12] = (__bf16)f3.x; a[13] = (__bf16)f3.y; a[14] = (__bf16)f3.z; a[15] = (__bf16)f3.w;

        // ---- B fragments: contiguous 32B per lane from packed weights
        const int off = c * (32 * 16);      // elements per k-chunk per n-tile
        v16bf b0 = *(const v16bf*)(pb0 + off);
        v16bf b1 = *(const v16bf*)(pb1 + off);
        v16bf b2 = *(const v16bf*)(pb2 + off);
        v16bf b3 = *(const v16bf*)(pb3 + off);

        acc0 = __builtin_amdgcn_wmma_f32_16x16x32_bf16(false, a, false, b0, (short)0, acc0, false, false);
        acc1 = __builtin_amdgcn_wmma_f32_16x16x32_bf16(false, a, false, b1, (short)0, acc1, false, false);
        acc2 = __builtin_amdgcn_wmma_f32_16x16x32_bf16(false, a, false, b2, (short)0, acc2, false, false);
        acc3 = __builtin_amdgcn_wmma_f32_16x16x32_bf16(false, a, false, b3, (short)0, acc3, false, false);
    }

    float beta = 0.0f;
    if (POST == 2) beta = 1.0f / (1.0f + expf(-skipPtr[0]));

#define EPILOGUE(T, ACC)                                                               \
    {                                                                                  \
        int   col  = ((nt0 + (T)) << 4) + row;                                         \
        float bval = bias ? bias[col] : 0.0f;                                          \
        _Pragma("unroll")                                                              \
        for (int r = 0; r < 8; ++r) {                                                  \
            int   m = m0 + r + (grp << 3);                                             \
            float v = (ACC)[r] + bval;                                                 \
            if (POST == 1)      v = (v >= 0.0f) ? v : 0.01f * v;                       \
            else if (POST == 2) v = beta * v + (1.0f - beta) * residual[(size_t)m * N + col]; \
            C[(size_t)m * N + col] = v;                                                \
        }                                                                              \
    }
    EPILOGUE(0, acc0)
    EPILOGUE(1, acc1)
    EPILOGUE(2, acc2)
    EPILOGUE(3, acc3)
#undef EPILOGUE
}

// elementwise exact GELU, in place, float4 vectorized (n % 4 == 0)
__global__ void gelu_inplace_kernel(float* __restrict__ x, int n4)
{
    int i = (int)(blockIdx.x * blockDim.x + threadIdx.x);
    if (i >= n4) return;
    float4 v = ((float4*)x)[i];
    v.x = 0.5f * v.x * (1.0f + erff(v.x * 0.70710678118654752f));
    v.y = 0.5f * v.y * (1.0f + erff(v.y * 0.70710678118654752f));
    v.z = 0.5f * v.z * (1.0f + erff(v.z * 0.70710678118654752f));
    v.w = 0.5f * v.w * (1.0f + erff(v.w * 0.70710678118654752f));
    ((float4*)x)[i] = v;
}

// init segment-max (order-preserving uint encoding of -inf) and denom=0
__global__ void init_seg_kernel(unsigned* __restrict__ amaxU, float* __restrict__ denom, int n)
{
    int i = (int)(blockIdx.x * blockDim.x + threadIdx.x);
    if (i < n) { amaxU[i] = 0x007FFFFFu; denom[i] = 0.0f; }
}

// alpha[e] = dot(q[dst], kr[src]) * p / sqrt(128);  segment-max via encoded atomicMax.
__global__ __launch_bounds__(256) void edge_alpha_kernel(
    const float* __restrict__ q, const float* __restrict__ kr,
    const int* __restrict__ esrc, const int* __restrict__ edst,
    const float* __restrict__ p, float* __restrict__ alpha,
    unsigned* __restrict__ amaxU, int E)
{
    int wave = (int)((blockIdx.x * blockDim.x + threadIdx.x) >> 5);
    int lane = (int)(threadIdx.x & 31);
    if (wave >= E) return;
    int s = esrc[wave], d = edst[wave];
    float4 qa = ((const float4*)(q  + (size_t)d * HID))[lane];
    float4 ka = ((const float4*)(kr + (size_t)s * HID))[lane];
    float sum = qa.x * ka.x + qa.y * ka.y + qa.z * ka.z + qa.w * ka.w;
#pragma unroll
    for (int off = 16; off > 0; off >>= 1) sum += __shfl_xor(sum, off, 32);
    if (lane == 0) {
        float a = sum * p[0] * 0.08838834764831845f;   // 1/sqrt(128)
        alpha[wave] = a;
        int bits = __float_as_int(a);
        unsigned enc = (unsigned)bits ^ ((unsigned)(bits >> 31) | 0x80000000u);
        atomicMax(amaxU + d, enc);
    }
}

// ex = exp(alpha - amax[dst]); denom[dst] += ex  (ex overwrites alpha)
__global__ void edge_expdenom_kernel(
    float* __restrict__ alpha, const unsigned* __restrict__ amaxU,
    float* __restrict__ denom, const int* __restrict__ edst, int E)
{
    int e = (int)(blockIdx.x * blockDim.x + threadIdx.x);
    if (e >= E) return;
    int d = edst[e];
    unsigned u = amaxU[d];
    int bits = (u & 0x80000000u) ? (int)(u ^ 0x80000000u) : (int)~u;
    float ex = expf(alpha[e] - __int_as_float(bits));
    alpha[e] = ex;
    atomicAdd(denom + d, ex);
}

// out[dst] += vr[src] * (ex / denom[dst]);  one wave per edge, float4 per lane
__global__ __launch_bounds__(256) void edge_scatter_kernel(
    const float* __restrict__ vr, const float* __restrict__ ex,
    const float* __restrict__ denom, const int* __restrict__ esrc,
    const int* __restrict__ edst, float* __restrict__ out, int E)
{
    int wave = (int)((blockIdx.x * blockDim.x + threadIdx.x) >> 5);
    int lane = (int)(threadIdx.x & 31);
    if (wave >= E) return;
    int s = esrc[wave], d = edst[wave];
    float w = ex[wave] / denom[d];
    float4 a = ((const float4*)(vr + (size_t)s * HID))[lane];
    float* op = out + (size_t)d * HID + lane * 4;
    atomicAdd(op + 0, a.x * w);
    atomicAdd(op + 1, a.y * w);
    atomicAdd(op + 2, a.z * w);
    atomicAdd(op + 3, a.w * w);
}

// final head: out[n,0:2] = h64[n,:] @ Wo2[64x2] + bo2 ; one wave per node
__global__ __launch_bounds__(256) void head2_kernel(
    const float* __restrict__ h, const float* __restrict__ Wo2,
    const float* __restrict__ bo2, float* __restrict__ out, int Nn)
{
    int wave = (int)((blockIdx.x * blockDim.x + threadIdx.x) >> 5);
    int lane = (int)(threadIdx.x & 31);
    if (wave >= Nn) return;
    float h0 = h[(size_t)wave * 64 + lane];
    float h1 = h[(size_t)wave * 64 + lane + 32];
    float a0 = h0 * Wo2[lane * 2 + 0] + h1 * Wo2[(lane + 32) * 2 + 0];
    float a1 = h0 * Wo2[lane * 2 + 1] + h1 * Wo2[(lane + 32) * 2 + 1];
#pragma unroll
    for (int off = 16; off > 0; off >>= 1) {
        a0 += __shfl_xor(a0, off, 32);
        a1 += __shfl_xor(a1, off, 32);
    }
    if (lane == 0) {
        out[(size_t)wave * 2 + 0] = a0 + bo2[0];
        out[(size_t)wave * 2 + 1] = a1 + bo2[1];
    }
}

// ---------------------------------------------------------------------------
static inline void launch_gemm(const float* A, const __bf16* P, const float* bias,
                               float* C, int M, int K, int N, int postMode,
                               const float* residual, const float* skipPtr, hipStream_t s)
{
    int tiles  = (M / 16) * (N / 64);
    int blocks = (tiles + 7) / 8;           // 8 waves / 256-thread block
    if (K == F_IN_) {                       // KCH = 24 (only POST=1 used)
        wmma_gemm_kernel<1, 24><<<blocks, 256, 0, s>>>(A, P, bias, C, M, N, residual, skipPtr);
    } else {                                // K = 128, KCH = 4
        switch (postMode) {
        case 1:
            wmma_gemm_kernel<1, 4><<<blocks, 256, 0, s>>>(A, P, bias, C, M, N, residual, skipPtr);
            break;
        case 2:
            wmma_gemm_kernel<2, 4><<<blocks, 256, 0, s>>>(A, P, bias, C, M, N, residual, skipPtr);
            break;
        default:
            wmma_gemm_kernel<0, 4><<<blocks, 256, 0, s>>>(A, P, bias, C, M, N, residual, skipPtr);
            break;
        }
    }
}

static inline void launch_pack(const float* W, __bf16* P, int K, int N, hipStream_t st)
{
    pack_w_kernel<<<(K * N + 255) / 256, 256, 0, st>>>(W, P, K, N);
}

extern "C" void kernel_launch(void* const* d_in, const int* in_sizes, int n_in,
                              void* d_out, int out_size, void* d_ws, size_t ws_size,
                              hipStream_t stream)
{
    (void)in_sizes; (void)n_in; (void)out_size; (void)ws_size;

    const float* features = (const float*)d_in[0];
    const int*   e0       = (const int*)d_in[1];   // [2,E] row0=src row1=dst
    const int*   e1       = (const int*)d_in[2];
    const float* W1  = (const float*)d_in[3];  const float* b1  = (const float*)d_in[4];
    const float* Wk  = (const float*)d_in[5];  const float* bk  = (const float*)d_in[6];
    const float* Wq  = (const float*)d_in[7];  const float* bq  = (const float*)d_in[8];
    const float* Wv  = (const float*)d_in[9];  const float* bv  = (const float*)d_in[10];
    const float* A0  = (const float*)d_in[11]; const float* M0  = (const float*)d_in[12];
    const float* p0  = (const float*)d_in[13];
    const float* A1  = (const float*)d_in[14]; const float* M1  = (const float*)d_in[15];
    const float* p1  = (const float*)d_in[16];
    const float* Wa  = (const float*)d_in[17]; const float* ba  = (const float*)d_in[18];
    const float* skip= (const float*)d_in[19];
    const float* Wo1 = (const float*)d_in[20]; const float* bo1 = (const float*)d_in[21];
    const float* Wo2 = (const float*)d_in[22]; const float* bo2 = (const float*)d_in[23];

    // ---- workspace carve-up (256B aligned) --------------------------------
    char* w = (char*)d_ws;
    auto alloc = [&](size_t bytes) -> void* {
        void* p = (void*)w;
        w += (bytes + 255) & ~(size_t)255;
        return p;
    };
    const size_t NB = (size_t)N_NODES * HID * sizeof(float);   // 25.6 MB
    float* x0   = (float*)alloc(NB);
    float* x1   = (float*)alloc(NB);
    float* kbuf = (float*)alloc(NB);
    float* qbuf = (float*)alloc(NB);
    float* vbuf = (float*)alloc(NB);
    float* kr   = (float*)alloc(NB);
    float* vr   = (float*)alloc(NB);
    float* oacc = (float*)alloc(NB);
    float*    alpha = (float*)alloc((size_t)N_EDGES * sizeof(float));
    unsigned* amaxU = (unsigned*)alloc((size_t)N_NODES * sizeof(unsigned));
    float*    denom = (float*)alloc((size_t)N_NODES * sizeof(float));
    __bf16* W1b  = (__bf16*)alloc((size_t)F_IN_ * HID * 2);
    __bf16* Wkb  = (__bf16*)alloc((size_t)HID * HID * 2);
    __bf16* Wqb  = (__bf16*)alloc((size_t)HID * HID * 2);
    __bf16* Wvb  = (__bf16*)alloc((size_t)HID * HID * 2);
    __bf16* A0b  = (__bf16*)alloc((size_t)HID * HID * 2);
    __bf16* M0b  = (__bf16*)alloc((size_t)HID * HID * 2);
    __bf16* A1b  = (__bf16*)alloc((size_t)HID * HID * 2);
    __bf16* M1b  = (__bf16*)alloc((size_t)HID * HID * 2);
    __bf16* Wab  = (__bf16*)alloc((size_t)HID * HID * 2);
    __bf16* Wo1b = (__bf16*)alloc((size_t)HID * 64 * 2);
    float* h64 = kr;   // reuse (free after layers)

    // ---- weights -> packed fragment-major bf16 ----------------------------
    launch_pack(W1, W1b, F_IN_, HID, stream);
    launch_pack(Wk, Wkb, HID, HID, stream);
    launch_pack(Wq, Wqb, HID, HID, stream);
    launch_pack(Wv, Wvb, HID, HID, stream);
    launch_pack(A0, A0b, HID, HID, stream);
    launch_pack(M0, M0b, HID, HID, stream);
    launch_pack(A1, A1b, HID, HID, stream);
    launch_pack(M1, M1b, HID, HID, stream);
    launch_pack(Wa, Wab, HID, HID, stream);
    launch_pack(Wo1, Wo1b, HID, 64, stream);

    // ---- linear1 + LeakyReLU ----------------------------------------------
    launch_gemm(features, W1b, b1, x0, N_NODES, F_IN_, HID, 1, nullptr, nullptr, stream);

    // ---- two HGT layers ----------------------------------------------------
    const int edgeWBlocks = (N_EDGES * 32 + 255) / 256;   // wave-per-edge kernels
    const int edgeTBlocks = (N_EDGES + 255) / 256;
    const int nodeBlocks  = (N_NODES + 255) / 256;

    const __bf16* Ab[2] = {A0b, A1b};
    const __bf16* Mb[2] = {M0b, M1b};
    const float*  pp[2] = {p0, p1};
    const int*    ee[2] = {e0, e1};

    for (int layer = 0; layer < 2; ++layer) {
        const float* xin = (layer == 0) ? x0 : x1;
        float*       xout = (layer == 0) ? x1 : x0;

        launch_gemm(xin, Wkb, bk, kbuf, N_NODES, HID, HID, 0, nullptr, nullptr, stream);
        launch_gemm(xin, Wqb, bq, qbuf, N_NODES, HID, HID, 0, nullptr, nullptr, stream);
        launch_gemm(xin, Wvb, bv, vbuf, N_NODES, HID, HID, 0, nullptr, nullptr, stream);
        hipMemsetAsync(oacc, 0, NB, stream);

        for (int rel = 0; rel < 2; ++rel) {
            const int* esrc = ee[rel];
            const int* edst = ee[rel] + N_EDGES;
            launch_gemm(kbuf, Ab[rel], nullptr, kr, N_NODES, HID, HID, 0, nullptr, nullptr, stream);
            launch_gemm(vbuf, Mb[rel], nullptr, vr, N_NODES, HID, HID, 0, nullptr, nullptr, stream);
            init_seg_kernel<<<nodeBlocks, 256, 0, stream>>>(amaxU, denom, N_NODES);
            edge_alpha_kernel<<<edgeWBlocks, 256, 0, stream>>>(qbuf, kr, esrc, edst,
                                                               pp[rel], alpha, amaxU, N_EDGES);
            edge_expdenom_kernel<<<edgeTBlocks, 256, 0, stream>>>(alpha, amaxU, denom, edst, N_EDGES);
            edge_scatter_kernel<<<edgeWBlocks, 256, 0, stream>>>(vr, alpha, denom, esrc, edst,
                                                                 oacc, N_EDGES);
        }
        // oacc = gelu(oacc); xout = beta*(oacc @ Wa + ba) + (1-beta)*xin
        gelu_inplace_kernel<<<(N_NODES * HID / 4 + 255) / 256, 256, 0, stream>>>(oacc, N_NODES * HID / 4);
        launch_gemm(oacc, Wab, ba, xout, N_NODES, HID, HID, 2, xin, skip, stream);
    }

    // ---- head: leaky(x @ Wo1 + bo1) @ Wo2 + bo2 ---------------------------
    launch_gemm(x0, Wo1b, bo1, h64, N_NODES, HID, 64, 1, nullptr, nullptr, stream);
    head2_kernel<<<(N_NODES * 32 + 255) / 256, 256, 0, stream>>>(h64, Wo2, bo2,
                                                                 (float*)d_out, N_NODES);
}
